// net_3882650435790
// MI455X (gfx1250) — matrix operand order repaired and checked by reading, not compiled, not run on hardware
//
#include <hip/hip_runtime.h>

typedef __attribute__((ext_vector_type(16))) _Float16 v16h;
typedef __attribute__((ext_vector_type(8)))  float    v8f;

#define L2EPS 1e-12f
#define MLP_WAVES 4

__device__ __forceinline__ float half_group_reduce(float v) {
  // sum across the 16-lane half-wave groups (xor masks stay within group)
  v += __shfl_xor(v, 1, 32);
  v += __shfl_xor(v, 2, 32);
  v += __shfl_xor(v, 4, 32);
  v += __shfl_xor(v, 8, 32);
  return v;
}

// ---------------- conv1: x[N,16] -> h1[N,32], one lane per (edge, d) ----------------
__global__ __launch_bounds__(256) void conv1_kernel(
    const float* __restrict__ x, const float* __restrict__ Kw,
    const long long* __restrict__ ei, float* __restrict__ h1, int E) {
  long long t = (long long)blockIdx.x * blockDim.x + threadIdx.x;
  long long e = t >> 4;
  int d = (int)(t & 15);
  if (e >= E) return;
  long long s  = ei[e];
  long long dn = ei[(long long)E + e];
  float xv = x[s * 16 + d];
  float k0 = Kw[e];
  float k1 = Kw[(long long)E + e];
  unsafeAtomicAdd(&h1[dn * 32 + d], k0 * xv);
  unsafeAtomicAdd(&h1[dn * 32 + 16 + d], k1 * xv);
}

// ---------------- conv2: g1[N,32] -> h2[N,64], one lane per (edge, c) ----------------
__global__ __launch_bounds__(256) void conv2_kernel(
    const float* __restrict__ g1, const float* __restrict__ Kw,
    const long long* __restrict__ ei, float* __restrict__ h2, int E) {
  long long t = (long long)blockIdx.x * blockDim.x + threadIdx.x;
  long long e = t >> 5;
  int c = (int)(t & 31);
  if (e >= E) return;
  long long s  = ei[e];
  long long dn = ei[(long long)E + e];
  float xv = g1[s * 32 + c];
  float k0 = Kw[e];
  float k1 = Kw[(long long)E + e];
  unsafeAtomicAdd(&h2[dn * 64 + c], k0 * xv);
  unsafeAtomicAdd(&h2[dn * 64 + 32 + c], k1 * xv);
}

// ------------- mlp1: g1 = l2norm(relu(h1[16x32] @ W1[32x32] + b1)) per 16-node tile -------------
__global__ __launch_bounds__(MLP_WAVES * 32) void mlp1_kernel(
    const float* __restrict__ h1, const float* __restrict__ W1,
    const float* __restrict__ b1, float* __restrict__ g1, int N) {
  int wave = blockIdx.x * MLP_WAVES + (threadIdx.x >> 5);
  int lane = threadIdx.x & 31;
  int ntiles = N / 16;
  if (wave >= ntiles) return;           // wave-uniform: EXEC stays all-ones
  int m  = lane & 15;                   // row for A, column for B/C
  int kh = lane >> 4;                   // K half selector
  long long row = (long long)wave * 16 + m;

  // A layout (16-bit 16x32): lanes 0-15 K=[0..7|16..23], lanes 16-31 K=[8..15|24..31]
  v16h a;
  const float* pa = h1 + row * 32;
#pragma unroll
  for (int i = 0; i < 8; ++i) {
    a[i]     = (_Float16)pa[kh * 8 + i];
    a[8 + i] = (_Float16)pa[16 + kh * 8 + i];
  }

  int kb = kh * 16;                     // B layout: slot t -> k = (t&8)*2 + kb/2 + (t&7)
  float out[2][8];
#pragma unroll
  for (int nt = 0; nt < 2; ++nt) {
    int n = nt * 16 + m;
    v16h b;
#pragma unroll
    for (int t = 0; t < 16; ++t) b[t] = (_Float16)W1[((t & 8) * 2 + kb / 2 + (t & 7)) * 32 + n];
    v8f c = {};
    c = __builtin_amdgcn_wmma_f32_16x16x32_f16(false, a, false, b, (short)0, c, false, false);
    float bias = b1[n];
#pragma unroll
    for (int r = 0; r < 8; ++r) {
      float v = c[r] + bias;
      out[nt][r] = v > 0.f ? v : 0.f;
    }
  }
  // L2 normalize each row (row mr = r + 8*kh lives across the 16-lane group)
#pragma unroll
  for (int r = 0; r < 8; ++r) {
    float s = out[0][r] * out[0][r] + out[1][r] * out[1][r];
    s = half_group_reduce(s);
    float inv = 1.f / fmaxf(sqrtf(s), L2EPS);
    long long orow = ((long long)wave * 16 + r + 8 * kh) * 32;
    g1[orow + m]      = out[0][r] * inv;
    g1[orow + 16 + m] = out[1][r] * inv;
  }
}

// ------- mlp2: out = l2norm( relu(h2[16x64]@W2a[64x128]+b2a) @ W2b[128x64] + b2b ) -------
__global__ __launch_bounds__(MLP_WAVES * 32) void mlp2_kernel(
    const float* __restrict__ h2,
    const float* __restrict__ W2a, const float* __restrict__ b2a,
    const float* __restrict__ W2b, const float* __restrict__ b2b,
    float* __restrict__ out, int N) {
  __shared__ float hid[MLP_WAVES][16][132];   // per-wave C->A transpose staging
  int wslot = threadIdx.x >> 5;
  int wave  = blockIdx.x * MLP_WAVES + wslot;
  int lane  = threadIdx.x & 31;
  int ntiles = N / 16;
  if (wave >= ntiles) return;           // wave-uniform
  int m  = lane & 15;
  int kh = lane >> 4;
  int kb = kh * 16;
  long long row = (long long)wave * 16 + m;

  // A (K=64) -> two K-steps of 32
  v16h a0, a1;
  const float* pa = h2 + row * 64;
#pragma unroll
  for (int i = 0; i < 8; ++i) {
    a0[i]     = (_Float16)pa[kh * 8 + i];
    a0[8 + i] = (_Float16)pa[16 + kh * 8 + i];
    a1[i]     = (_Float16)pa[32 + kh * 8 + i];
    a1[8 + i] = (_Float16)pa[48 + kh * 8 + i];
  }

  // GEMM1: hid[16x128] = relu(A @ W2a + b2a), written to LDS in row/col order
#pragma unroll
  for (int nt = 0; nt < 8; ++nt) {
    int n = nt * 16 + m;
    v16h b0, b1v;
#pragma unroll
    for (int t = 0; t < 16; ++t) {
      b0[t]  = (_Float16)W2a[((t & 8) * 2 + kb / 2 + (t & 7)) * 128 + n];
      b1v[t] = (_Float16)W2a[(32 + (t & 8) * 2 + kb / 2 + (t & 7)) * 128 + n];
    }
    v8f c = {};
    c = __builtin_amdgcn_wmma_f32_16x16x32_f16(false, a0, false, b0,  (short)0, c, false, false);
    c = __builtin_amdgcn_wmma_f32_16x16x32_f16(false, a1, false, b1v, (short)0, c, false, false);
    float bias = b2a[n];
#pragma unroll
    for (int r = 0; r < 8; ++r) {
      float v = c[r] + bias;
      hid[wslot][r + 8 * kh][n] = v > 0.f ? v : 0.f;
    }
  }
  // intra-wave LDS RAW: DS ops are in-order per wave; wait + compiler barrier for safety
  asm volatile("s_wait_dscnt 0" ::: "memory");

  // GEMM2: out16x64 = hid @ W2b, K=128 in four K-steps
  v8f c2[4];
#pragma unroll
  for (int nt = 0; nt < 4; ++nt) { v8f z = {}; c2[nt] = z; }
#pragma unroll
  for (int ks = 0; ks < 4; ++ks) {
    v16h a;
    const float* pl = &hid[wslot][m][ks * 32];
#pragma unroll
    for (int i = 0; i < 8; ++i) {
      a[i]     = (_Float16)pl[kh * 8 + i];
      a[8 + i] = (_Float16)pl[16 + kh * 8 + i];
    }
#pragma unroll
    for (int nt = 0; nt < 4; ++nt) {
      int n = nt * 16 + m;
      v16h b;
#pragma unroll
      for (int t = 0; t < 16; ++t)
        b[t] = (_Float16)W2b[(ks * 32 + (t & 8) * 2 + kb / 2 + (t & 7)) * 64 + n];
      c2[nt] = __builtin_amdgcn_wmma_f32_16x16x32_f16(false, a, false, b, (short)0, c2[nt], false, false);
    }
  }
  // bias + L2 norm over 64 channels + store
  float vals[4][8];
#pragma unroll
  for (int nt = 0; nt < 4; ++nt) {
    float bias = b2b[nt * 16 + m];
#pragma unroll
    for (int r = 0; r < 8; ++r) vals[nt][r] = c2[nt][r] + bias;
  }
#pragma unroll
  for (int r = 0; r < 8; ++r) {
    float s = 0.f;
#pragma unroll
    for (int nt = 0; nt < 4; ++nt) s += vals[nt][r] * vals[nt][r];
    s = half_group_reduce(s);
    float inv = 1.f / fmaxf(sqrtf(s), L2EPS);
    long long orow = ((long long)wave * 16 + r + 8 * kh) * 64;
#pragma unroll
    for (int nt = 0; nt < 4; ++nt)
      out[orow + nt * 16 + m] = vals[nt][r] * inv;
  }
}

extern "C" void kernel_launch(void* const* d_in, const int* in_sizes, int n_in,
                              void* d_out, int out_size, void* d_ws, size_t ws_size,
                              hipStream_t stream) {
  const float*     x   = (const float*)d_in[0];
  const float*     Kw  = (const float*)d_in[1];
  const long long* ei  = (const long long*)d_in[2];
  const float*     W1  = (const float*)d_in[3];
  const float*     b1  = (const float*)d_in[4];
  const float*     W2a = (const float*)d_in[5];
  const float*     b2a = (const float*)d_in[6];
  const float*     W2b = (const float*)d_in[7];
  const float*     b2b = (const float*)d_in[8];
  float* outp = (float*)d_out;

  int N = in_sizes[0] / 16;   // 50000
  int E = in_sizes[1] / 2;    // 1.6M

  float* h1 = (float*)d_ws;                  // N*32 floats
  float* g1 = h1 + (size_t)N * 32;           // N*32 floats
  float* h2 = g1 + (size_t)N * 32;           // N*64 floats
  hipMemsetAsync(h1, 0, (size_t)N * 32 * sizeof(float), stream);
  hipMemsetAsync(h2, 0, (size_t)N * 64 * sizeof(float), stream);

  {
    long long threads = (long long)E * 16;
    unsigned grid = (unsigned)((threads + 255) / 256);
    conv1_kernel<<<grid, 256, 0, stream>>>(x, Kw, ei, h1, E);
  }
  {
    int ntiles = N / 16;
    int grid = (ntiles + MLP_WAVES - 1) / MLP_WAVES;
    mlp1_kernel<<<grid, MLP_WAVES * 32, 0, stream>>>(h1, W1, b1, g1, N);
  }
  {
    long long threads = (long long)E * 32;
    unsigned grid = (unsigned)((threads + 255) / 256);
    conv2_kernel<<<grid, 256, 0, stream>>>(g1, Kw, ei, h2, E);
  }
  {
    int ntiles = N / 16;
    int grid = (ntiles + MLP_WAVES - 1) / MLP_WAVES;
    mlp2_kernel<<<grid, MLP_WAVES * 32, 0, stream>>>(h2, W2a, b2a, W2b, b2b, outp, N);
  }
}
